// GRAPEPatternAware_3049426780538
// MI455X (gfx1250) — matrix operand order repaired and checked by reading, not compiled, not run on hardware
//
#include <hip/hip_runtime.h>
#include <hip/hip_bf16.h>

// ---------------------------------------------------------------------------
// GRAPE pattern-aware GNN for MI455X (gfx1250, wave32, WMMA f16->f32)
// Block tile: 64 rows x 256 cols, 4 waves; each wave: 4 row-tiles x 4 n-tiles
// = 16 WMMAs per K-step on 8 loaded fragments (4 A + 4 B).
// ---------------------------------------------------------------------------

#define G_     2048
#define F_     64
#define H_     256
#define C_     10
#define L_     3
#define NNODE  (G_ * (F_ + 1))   // 133120 = 64 * 2080
#define E_     262144

typedef _Float16 half_t;
typedef __attribute__((ext_vector_type(16))) _Float16 v16h;
typedef __attribute__((ext_vector_type(8)))  _Float16 v8h;
typedef __attribute__((ext_vector_type(8)))  float    v8f;

__device__ __forceinline__ v8f zero8() {
    v8f z = {0.f, 0.f, 0.f, 0.f, 0.f, 0.f, 0.f, 0.f};
    return z;
}

__device__ __forceinline__ v16h combine16(v8h lo, v8h hi8) {
    v16h a;
#pragma unroll
    for (int i = 0; i < 8; ++i) { a[i] = lo[i]; a[i + 8] = hi8[i]; }
    return a;
}

// A fragment (16x32 f16): lane m = row, hi selects K-chunk pair {kb+8hi, kb+16+8hi}
__device__ __forceinline__ v16h load_a(const half_t* __restrict__ row, int kb, int hi) {
    v8h c0 = *(const v8h*)(row + kb + hi * 8);
    v8h c1 = *(const v8h*)(row + kb + 16 + hi * 8);
    return combine16(c0, c1);
}

__device__ __forceinline__ v8f wmma_f16(v16h a, v16h b, v8f c) {
    return __builtin_amdgcn_wmma_f32_16x16x32_f16(
        /*neg_a=*/false, a, /*neg_b=*/false, b,
        /*c_mod=*/(short)0, c, /*reuse_a=*/false, /*reuse_b=*/false);
}

__device__ __forceinline__ v16h load_b(const half_t* __restrict__ Bp, int kt, int nt, int lane) {
    return *(const v16h*)(Bp + (((size_t)kt * 16 + nt) * 32 + lane) * 16);
}

// ---------------------------------------------------------------------------
// Weight packing: W[K][N] f32 row-major -> WMMA B-fragment layout f16.
// lane l holds column n = nt*16 + (l&15), K = kt*32 + (i>>3)*16 + (l>>4)*8 + (i&7), i=0..15
// ---------------------------------------------------------------------------
__global__ __launch_bounds__(256) void pack_w(const float* __restrict__ W,
                                              half_t* __restrict__ out,
                                              int K, int N) {
    int NT = N >> 4;
    size_t total = (size_t)K * N;
    size_t o = (size_t)blockIdx.x * 256 + threadIdx.x;
    if (o >= total) return;
    int i  = (int)(o & 15);
    int l  = (int)((o >> 4) & 31);
    size_t t = o >> 9;                // tile index = kt*NT + nt
    int nt = (int)(t % NT);
    int kt = (int)(t / NT);
    int k  = kt * 32 + (i >> 3) * 16 + (l >> 4) * 8 + (i & 7);
    int n  = nt * 16 + (l & 15);
    out[o] = (half_t)W[(size_t)k * N + n];
}

__global__ __launch_bounds__(256) void cvt_f32_f16(const float* __restrict__ in,
                                                   half_t* __restrict__ out,
                                                   size_t n) {
    size_t i = (size_t)blockIdx.x * 256 + threadIdx.x;
    if (i < n) out[i] = (half_t)in[i];
}

__global__ __launch_bounds__(256) void count_edges(const int* __restrict__ dst,
                                                   float* __restrict__ cnt) {
    int i = blockIdx.x * 256 + threadIdx.x;
    if (i < E_) atomicAdd(&cnt[dst[i]], 1.0f);
}

// ---------------------------------------------------------------------------
// Generic GEMM: out16[M,256] = act( A[M,K] f16 @ Bp(packed Kx256) + bias )
// ---------------------------------------------------------------------------
__global__ __launch_bounds__(128) void gemm_rm(const half_t* __restrict__ A,
                                               const half_t* __restrict__ Bp,
                                               const float* __restrict__ bias,
                                               half_t* __restrict__ out16,
                                               int K, int relu) {
    const int lane = threadIdx.x & 31;
    const int wave = threadIdx.x >> 5;
    const int row0 = blockIdx.x << 6;
    const int mrow = lane & 15;
    const int hi   = lane >> 4;

    const half_t* arow[4];
#pragma unroll
    for (int rt = 0; rt < 4; ++rt)
        arow[rt] = A + (size_t)(row0 + rt * 16 + mrow) * K;

    v8f acc[4][4];
#pragma unroll
    for (int rt = 0; rt < 4; ++rt)
#pragma unroll
        for (int j = 0; j < 4; ++j) acc[rt][j] = zero8();

    const int nkt = K >> 5;
    for (int kt = 0; kt < nkt; ++kt) {
        const int kb = kt << 5;
        v16h a[4];
#pragma unroll
        for (int rt = 0; rt < 4; ++rt) a[rt] = load_a(arow[rt], kb, hi);
        v16h b[4];
#pragma unroll
        for (int j = 0; j < 4; ++j) b[j] = load_b(Bp, kt, (wave << 2) + j, lane);
#pragma unroll
        for (int rt = 0; rt < 4; ++rt)
#pragma unroll
            for (int j = 0; j < 4; ++j)
                acc[rt][j] = wmma_f16(a[rt], b[j], acc[rt][j]);
    }

#pragma unroll
    for (int j = 0; j < 4; ++j) {
        int col = (((wave << 2) + j) << 4) + (lane & 15);
        float bv = bias[col];
#pragma unroll
        for (int rt = 0; rt < 4; ++rt)
#pragma unroll
            for (int r = 0; r < 8; ++r) {
                int row = row0 + rt * 16 + r + (hi << 3);
                float v = acc[rt][j][r] + bv;
                if (relu) v = fmaxf(v, 0.f);
                out16[(size_t)row * H_ + col] = (half_t)v;
            }
    }
}

// ---------------------------------------------------------------------------
// Fused message layer, 64 edges per block:
//   t = relu([h[src] || (ea*We+be)] @ Wm1 + bm1)   (K=512, -> 32KB LDS)
//   m = t @ Wm2 + bm2                              (K=256, <- LDS)
//   atomicAdd(agg[dst], m)
// ---------------------------------------------------------------------------
__global__ __launch_bounds__(128) void msg_layer(const half_t* __restrict__ hbuf,
                                                 const float* __restrict__ ea,
                                                 const float* __restrict__ WeF,
                                                 const float* __restrict__ beF,
                                                 const int* __restrict__ srcIdx,
                                                 const int* __restrict__ dstIdx,
                                                 const half_t* __restrict__ pWm1,
                                                 const float* __restrict__ bm1,
                                                 const half_t* __restrict__ pWm2,
                                                 const float* __restrict__ bm2,
                                                 float* __restrict__ agg) {
    __shared__ half_t tls[64 * H_];   // 32 KB intermediate tile

    const int lane = threadIdx.x & 31;
    const int wave = threadIdx.x >> 5;
    const int e0   = blockIdx.x << 6;
    const int mrow = lane & 15;
    const int hi   = lane >> 4;

    const half_t* hrow[4];
    float eav[4];
#pragma unroll
    for (int rt = 0; rt < 4; ++rt) {
        int ed = e0 + rt * 16 + mrow;
        hrow[rt] = hbuf + (size_t)srcIdx[ed] * H_;
        eav[rt]  = ea[ed];
    }

    // ---- Phase 1: K = 512
    v8f acc[4][4];
#pragma unroll
    for (int rt = 0; rt < 4; ++rt)
#pragma unroll
        for (int j = 0; j < 4; ++j) acc[rt][j] = zero8();

    for (int kt = 0; kt < 16; ++kt) {
        const int kb = kt << 5;
        v16h a[4];
        if (kb < 256) {                       // uniform branch (kb step 32)
#pragma unroll
            for (int rt = 0; rt < 4; ++rt) a[rt] = load_a(hrow[rt], kb, hi);
        } else {
            int k0 = kb - 256 + hi * 8;
            float wv0[8], wv1[8], bv0[8], bv1[8];
#pragma unroll
            for (int i = 0; i < 8; ++i) {
                wv0[i] = WeF[k0 + i];      bv0[i] = beF[k0 + i];
                wv1[i] = WeF[k0 + 16 + i]; bv1[i] = beF[k0 + 16 + i];
            }
#pragma unroll
            for (int rt = 0; rt < 4; ++rt)
#pragma unroll
                for (int i = 0; i < 8; ++i) {
                    a[rt][i]     = (half_t)(eav[rt] * wv0[i] + bv0[i]);
                    a[rt][i + 8] = (half_t)(eav[rt] * wv1[i] + bv1[i]);
                }
        }
        v16h b[4];
#pragma unroll
        for (int j = 0; j < 4; ++j) b[j] = load_b(pWm1, kt, (wave << 2) + j, lane);
#pragma unroll
        for (int rt = 0; rt < 4; ++rt)
#pragma unroll
            for (int j = 0; j < 4; ++j)
                acc[rt][j] = wmma_f16(a[rt], b[j], acc[rt][j]);
    }

    // relu + stage into LDS
#pragma unroll
    for (int j = 0; j < 4; ++j) {
        int col = (((wave << 2) + j) << 4) + (lane & 15);
        float bv = bm1[col];
#pragma unroll
        for (int rt = 0; rt < 4; ++rt)
#pragma unroll
            for (int r = 0; r < 8; ++r) {
                int tr = rt * 16 + r + (hi << 3);
                tls[tr * H_ + col] = (half_t)fmaxf(acc[rt][j][r] + bv, 0.f);
            }
    }
    __syncthreads();

    // ---- Phase 2: K = 256 from LDS, then scatter-add into agg
    v8f acc2[4][4];
#pragma unroll
    for (int rt = 0; rt < 4; ++rt)
#pragma unroll
        for (int j = 0; j < 4; ++j) acc2[rt][j] = zero8();

    const half_t* trow[4];
#pragma unroll
    for (int rt = 0; rt < 4; ++rt) trow[rt] = tls + (rt * 16 + mrow) * H_;

    for (int kt = 0; kt < 8; ++kt) {
        const int kb = kt << 5;
        v16h a[4];
#pragma unroll
        for (int rt = 0; rt < 4; ++rt) a[rt] = load_a(trow[rt], kb, hi);
        v16h b[4];
#pragma unroll
        for (int j = 0; j < 4; ++j) b[j] = load_b(pWm2, kt, (wave << 2) + j, lane);
#pragma unroll
        for (int rt = 0; rt < 4; ++rt)
#pragma unroll
            for (int j = 0; j < 4; ++j)
                acc2[rt][j] = wmma_f16(a[rt], b[j], acc2[rt][j]);
    }

#pragma unroll
    for (int j = 0; j < 4; ++j) {
        int col = (((wave << 2) + j) << 4) + (lane & 15);
        float bv = bm2[col];
#pragma unroll
        for (int rt = 0; rt < 4; ++rt)
#pragma unroll
            for (int r = 0; r < 8; ++r) {
                int ed = e0 + rt * 16 + r + (hi << 3);
                int dn = dstIdx[ed];
                atomicAdd(&agg[(size_t)dn * H_ + col], acc2[rt][j][r] + bv);
            }
    }
}

// ---------------------------------------------------------------------------
// Update: h_out = relu([h || agg/max(cnt,1)] @ Wu + bu), K = 512
// ---------------------------------------------------------------------------
__global__ __launch_bounds__(128) void update_layer(const half_t* __restrict__ hin,
                                                    const float* __restrict__ agg,
                                                    const float* __restrict__ cnt,
                                                    const half_t* __restrict__ pWu,
                                                    const float* __restrict__ bu,
                                                    half_t* __restrict__ hout) {
    const int lane = threadIdx.x & 31;
    const int wave = threadIdx.x >> 5;
    const int row0 = blockIdx.x << 6;
    const int mrow = lane & 15;
    const int hi   = lane >> 4;

    const half_t* hrow[4];
    const float*  arow[4];
    float inv[4];
#pragma unroll
    for (int rt = 0; rt < 4; ++rt) {
        int row = row0 + rt * 16 + mrow;
        inv[rt]  = 1.0f / fmaxf(cnt[row], 1.0f);
        hrow[rt] = hin + (size_t)row * H_;
        arow[rt] = agg + (size_t)row * H_;
    }

    v8f acc[4][4];
#pragma unroll
    for (int rt = 0; rt < 4; ++rt)
#pragma unroll
        for (int j = 0; j < 4; ++j) acc[rt][j] = zero8();

    for (int kt = 0; kt < 16; ++kt) {
        const int kb = kt << 5;
        v16h a[4];
        if (kb < 256) {
#pragma unroll
            for (int rt = 0; rt < 4; ++rt) a[rt] = load_a(hrow[rt], kb, hi);
        } else {
            int k0 = kb - 256 + hi * 8;
#pragma unroll
            for (int rt = 0; rt < 4; ++rt)
#pragma unroll
                for (int i = 0; i < 8; ++i) {
                    a[rt][i]     = (half_t)(arow[rt][k0 + i] * inv[rt]);
                    a[rt][i + 8] = (half_t)(arow[rt][k0 + 16 + i] * inv[rt]);
                }
        }
        v16h b[4];
#pragma unroll
        for (int j = 0; j < 4; ++j) b[j] = load_b(pWu, kt, (wave << 2) + j, lane);
#pragma unroll
        for (int rt = 0; rt < 4; ++rt)
#pragma unroll
            for (int j = 0; j < 4; ++j)
                acc[rt][j] = wmma_f16(a[rt], b[j], acc[rt][j]);
    }

#pragma unroll
    for (int j = 0; j < 4; ++j) {
        int col = (((wave << 2) + j) << 4) + (lane & 15);
        float bv = bu[col];
#pragma unroll
        for (int rt = 0; rt < 4; ++rt)
#pragma unroll
            for (int r = 0; r < 8; ++r) {
                int row = row0 + rt * 16 + r + (hi << 3);
                hout[(size_t)row * H_ + col] = (half_t)fmaxf(acc[rt][j][r] + bv, 0.f);
            }
    }
}

// ---------------------------------------------------------------------------
// Final concat GEMM: zc = relu([h[g*65] || pat[g]] @ Wc1 + bc1), K=512, f32 out
// ---------------------------------------------------------------------------
__global__ __launch_bounds__(128) void final_concat(const half_t* __restrict__ hbuf,
                                                    const half_t* __restrict__ pat,
                                                    const half_t* __restrict__ pWc1,
                                                    const float* __restrict__ bc1,
                                                    float* __restrict__ zc) {
    const int lane = threadIdx.x & 31;
    const int wave = threadIdx.x >> 5;
    const int row0 = blockIdx.x << 6;
    const int mrow = lane & 15;
    const int hi   = lane >> 4;

    const half_t* hrow[4];
    const half_t* prow[4];
#pragma unroll
    for (int rt = 0; rt < 4; ++rt) {
        int g = row0 + rt * 16 + mrow;
        hrow[rt] = hbuf + (size_t)(g * (F_ + 1)) * H_;  // first node of graph g
        prow[rt] = pat + (size_t)g * H_;
    }

    v8f acc[4][4];
#pragma unroll
    for (int rt = 0; rt < 4; ++rt)
#pragma unroll
        for (int j = 0; j < 4; ++j) acc[rt][j] = zero8();

    for (int kt = 0; kt < 16; ++kt) {
        const int kb = kt << 5;
        v16h a[4];
        if (kb < 256) {
#pragma unroll
            for (int rt = 0; rt < 4; ++rt) a[rt] = load_a(hrow[rt], kb, hi);
        } else {
#pragma unroll
            for (int rt = 0; rt < 4; ++rt) a[rt] = load_a(prow[rt], kb - 256, hi);
        }
        v16h b[4];
#pragma unroll
        for (int j = 0; j < 4; ++j) b[j] = load_b(pWc1, kt, (wave << 2) + j, lane);
#pragma unroll
        for (int rt = 0; rt < 4; ++rt)
#pragma unroll
            for (int j = 0; j < 4; ++j)
                acc[rt][j] = wmma_f16(a[rt], b[j], acc[rt][j]);
    }

#pragma unroll
    for (int j = 0; j < 4; ++j) {
        int col = (((wave << 2) + j) << 4) + (lane & 15);
        float bv = bc1[col];
#pragma unroll
        for (int rt = 0; rt < 4; ++rt)
#pragma unroll
            for (int r = 0; r < 8; ++r) {
                int row = row0 + rt * 16 + r + (hi << 3);
                zc[(size_t)row * H_ + col] = fmaxf(acc[rt][j][r] + bv, 0.f);
            }
    }
}

// Skinny head: out[G,10] = zc[G,256] @ Wc2[256,10] + bc2 (VALU; 10 cols < 16)
__global__ __launch_bounds__(256) void head(const float* __restrict__ zc,
                                            const float* __restrict__ Wc2,
                                            const float* __restrict__ bc2,
                                            float* __restrict__ out) {
    int idx = blockIdx.x * 256 + threadIdx.x;
    if (idx >= G_ * C_) return;
    int g = idx / C_, c = idx % C_;
    float s = bc2[c];
    const float* zr = zc + (size_t)g * H_;
#pragma unroll 8
    for (int k = 0; k < H_; ++k) s += zr[k] * Wc2[k * C_ + c];
    out[idx] = s;
}

// ---------------------------------------------------------------------------
extern "C" void kernel_launch(void* const* d_in, const int* in_sizes, int n_in,
                              void* d_out, int out_size, void* d_ws, size_t ws_size,
                              hipStream_t stream) {
    const float* x    = (const float*)d_in[0];
    const float* ea   = (const float*)d_in[1];
    const float* mask = (const float*)d_in[2];
    const int*   eidx = (const int*)d_in[3];
    // d_in[4] = batch (unused: first node of graph g is at index g*65)
    const float* Wn  = (const float*)d_in[5];
    const float* bn  = (const float*)d_in[6];
    const float* We  = (const float*)d_in[7];
    const float* be  = (const float*)d_in[8];
    const float* Wm1 = (const float*)d_in[9];
    const float* bm1 = (const float*)d_in[10];
    const float* Wm2 = (const float*)d_in[11];
    const float* bm2 = (const float*)d_in[12];
    const float* Wu  = (const float*)d_in[13];
    const float* bu  = (const float*)d_in[14];
    const float* Wp1 = (const float*)d_in[15];
    const float* bp1 = (const float*)d_in[16];
    const float* Wp2 = (const float*)d_in[17];
    const float* bp2 = (const float*)d_in[18];
    const float* Wp3 = (const float*)d_in[19];
    const float* bp3 = (const float*)d_in[20];
    const float* Wc1 = (const float*)d_in[21];
    const float* bc1 = (const float*)d_in[22];
    const float* Wc2 = (const float*)d_in[23];
    const float* bc2 = (const float*)d_in[24];

    const int* srcIdx = eidx;
    const int* dstIdx = eidx + E_;

    // ---- workspace carve-up
    uint8_t* ws = (uint8_t*)d_ws;
    size_t off = 0;
    auto take = [&](size_t bytes) -> void* {
        void* p = ws + off;
        off += (bytes + 255) & ~(size_t)255;
        return p;
    };
    half_t* h_a  = (half_t*)take((size_t)NNODE * H_ * 2);
    half_t* h_b  = (half_t*)take((size_t)NNODE * H_ * 2);
    float*  agg  = (float*) take((size_t)NNODE * H_ * 4);
    float*  cnt  = (float*) take((size_t)NNODE * 4);
    half_t* x16  = (half_t*)take((size_t)NNODE * F_ * 2);
    half_t* m16  = (half_t*)take((size_t)G_ * F_ * 2);
    half_t* p_a  = (half_t*)take((size_t)G_ * H_ * 2);
    half_t* p_b  = (half_t*)take((size_t)G_ * H_ * 2);
    float*  zc   = (float*) take((size_t)G_ * H_ * 4);
    half_t* pWn  = (half_t*)take((size_t)F_ * H_ * 2);
    half_t* pWm1 = (half_t*)take((size_t)L_ * 2 * H_ * H_ * 2);
    half_t* pWm2 = (half_t*)take((size_t)L_ * H_ * H_ * 2);
    half_t* pWu  = (half_t*)take((size_t)L_ * 2 * H_ * H_ * 2);
    half_t* pWp1 = (half_t*)take((size_t)F_ * H_ * 2);
    half_t* pWp2 = (half_t*)take((size_t)H_ * H_ * 2);
    half_t* pWp3 = (half_t*)take((size_t)H_ * H_ * 2);
    half_t* pWc1 = (half_t*)take((size_t)2 * H_ * H_ * 2);
    (void)ws_size; (void)in_sizes; (void)n_in; (void)out_size;

    // ---- weight packing + input conversion
    pack_w<<<(F_ * H_ + 255) / 256, 256, 0, stream>>>(Wn, pWn, F_, H_);
    pack_w<<<(F_ * H_ + 255) / 256, 256, 0, stream>>>(Wp1, pWp1, F_, H_);
    pack_w<<<(H_ * H_ + 255) / 256, 256, 0, stream>>>(Wp2, pWp2, H_, H_);
    pack_w<<<(H_ * H_ + 255) / 256, 256, 0, stream>>>(Wp3, pWp3, H_, H_);
    pack_w<<<(2 * H_ * H_ + 255) / 256, 256, 0, stream>>>(Wc1, pWc1, 2 * H_, H_);
    for (int l = 0; l < L_; ++l) {
        size_t o2 = (size_t)l * 2 * H_ * H_;
        size_t o1 = (size_t)l * H_ * H_;
        pack_w<<<(2 * H_ * H_ + 255) / 256, 256, 0, stream>>>(Wm1 + o2, pWm1 + o2, 2 * H_, H_);
        pack_w<<<(H_ * H_ + 255) / 256, 256, 0, stream>>>(Wm2 + o1, pWm2 + o1, H_, H_);
        pack_w<<<(2 * H_ * H_ + 255) / 256, 256, 0, stream>>>(Wu + o2, pWu + o2, 2 * H_, H_);
    }
    {
        size_t n = (size_t)NNODE * F_;
        cvt_f32_f16<<<(unsigned)((n + 255) / 256), 256, 0, stream>>>(x, x16, n);
        size_t nm = (size_t)G_ * F_;
        cvt_f32_f16<<<(unsigned)((nm + 255) / 256), 256, 0, stream>>>(mask, m16, nm);
    }

    // ---- degree counts (scatter-mean denominator)
    hipMemsetAsync(cnt, 0, (size_t)NNODE * 4, stream);
    count_edges<<<E_ / 256, 256, 0, stream>>>(dstIdx, cnt);

    // ---- node projection: h = x @ Wn + bn
    gemm_rm<<<NNODE / 64, 128, 0, stream>>>(x16, pWn, bn, h_a, F_, /*relu=*/0);

    // ---- message passing layers
    half_t* h_cur = h_a;
    half_t* h_nxt = h_b;
    for (int l = 0; l < L_; ++l) {
        size_t o2 = (size_t)l * 2 * H_ * H_;
        size_t o1 = (size_t)l * H_ * H_;
        hipMemsetAsync(agg, 0, (size_t)NNODE * H_ * 4, stream);
        msg_layer<<<E_ / 64, 128, 0, stream>>>(h_cur, ea, We, be, srcIdx, dstIdx,
                                               pWm1 + o2, bm1 + (size_t)l * H_,
                                               pWm2 + o1, bm2 + (size_t)l * H_, agg);
        update_layer<<<NNODE / 64, 128, 0, stream>>>(h_cur, agg, cnt,
                                                     pWu + o2, bu + (size_t)l * H_, h_nxt);
        half_t* t = h_cur; h_cur = h_nxt; h_nxt = t;
    }

    // ---- pattern encoder: Lin-ReLU-Lin-ReLU-Lin
    gemm_rm<<<G_ / 64, 128, 0, stream>>>(m16, pWp1, bp1, p_a, F_, 1);
    gemm_rm<<<G_ / 64, 128, 0, stream>>>(p_a, pWp2, bp2, p_b, H_, 1);
    gemm_rm<<<G_ / 64, 128, 0, stream>>>(p_b, pWp3, bp3, p_a, H_, 0);

    // ---- classifier
    final_concat<<<G_ / 64, 128, 0, stream>>>(h_cur, p_a, pWc1, bc1, zc);
    head<<<(G_ * C_ + 255) / 256, 256, 0, stream>>>(zc, Wc2, bc2, (float*)d_out);
}
